// MambaModule_32736240730674
// MI455X (gfx1250) — matrix operand order repaired and checked by reading, not compiled, not run on hardware
//
#include <hip/hip_runtime.h>
#include <hip/hip_bf16.h>

// ---------------------------------------------------------------------------
// Mamba forward for MI455X (gfx1250, wave32).
// GEMMs use v_wmma_f32_16x16x32_bf16; selective scan is a register-state
// recurrence with async-to-LDS staged B/C; everything else is BW-bound VALU.
// ---------------------------------------------------------------------------

typedef __bf16 bf16_t;
typedef __attribute__((ext_vector_type(16))) __bf16 v16bf;
typedef __attribute__((ext_vector_type(8)))  __bf16 v8bf;
typedef __attribute__((ext_vector_type(8)))  float  v8f;
typedef int v4i_vs __attribute__((vector_size(16)));   // matches builtin param

#define BATCH    16
#define SEQ      4096
#define DMODEL   256
#define ED       512
#define NSTATE   16
#define DCONV    4
#define DTRANK   16
#define NLAYERS  4
#define MROWS    (BATCH * SEQ)          // 65536

__device__ __forceinline__ bf16_t to_bf16(float f) { return (bf16_t)f; }

// A-fragment (16x32 bf16): lane (l<16) row M=l; v0..v3 hold K = k0+8g .. +7,
// v4..v7 hold K = k0+16+8g .. +7 (g = lane>>4).  Two 16B loads.
__device__ __forceinline__ v16bf load_frag_a(const bf16_t* __restrict__ base,
                                             int K, int row, int k0g) {
  const v8bf* p = reinterpret_cast<const v8bf*>(base + (size_t)row * K + k0g);
  v8bf lo = p[0];
  v8bf hi = p[2];   // +16 elements
  v16bf f;
#pragma unroll
  for (int i = 0; i < 8; ++i) { f[i] = lo[i]; f[i + 8] = hi[i]; }
  return f;
}

// B-fragment (32x16 bf16, stored as Bt[N][K]): lane (l%16) = column N;
// v0..v3 hold K = k0+8g .. +7, v4..v7 hold K = k0+16+8g .. +7 (g = lane>>4).
__device__ __forceinline__ v16bf load_frag_b(const bf16_t* __restrict__ base,
                                             int K, int row, int k0, int g) {
  const v8bf* p = reinterpret_cast<const v8bf*>(base + (size_t)row * K + k0 + 8 * g);
  v8bf lo = p[0];
  v8bf hi = p[2];
  v16bf f;
#pragma unroll
  for (int i = 0; i < 8; ++i) { f[i] = lo[i]; f[i + 8] = hi[i]; }
  return f;
}

// ---------------------------------------------------------------------------
// bf16 WMMA GEMM:  C[M,N] = A[M,K] * B[K,N] (+ Resid), B given transposed
// as Bt[N,K].  Block = 256 thr = 8 waves arranged 4(M) x 2(N); each wave
// owns a 16x64 tile (4 accumulators, A fragment reused 4x per k-step).
// Out-of-range N sub-tiles load a clamped row (result discarded at store),
// so the inner loop is branch-free: clause(loads) -> 4x wmma.
// Requires: M % 64 == 0, K % 32 == 0, N % 16 == 0.
// ---------------------------------------------------------------------------
__global__ __launch_bounds__(256)
void gemm_bf16_wmma(const bf16_t* __restrict__ A, const bf16_t* __restrict__ Bt,
                    float* __restrict__ Cout, const float* __restrict__ Resid,
                    int M, int N, int K, int ldc) {
  const int lane = threadIdx.x & 31;
  const int wave = threadIdx.x >> 5;
  const int wr   = wave & 3;
  const int wc   = wave >> 2;
  const int g    = lane >> 4;
  const int l16  = lane & 15;

  const int tileM = blockIdx.y * 64 + wr * 16;
  const int tileN = blockIdx.x * 128 + wc * 64;
  const int arow  = tileM + l16;

  v8f acc[4] = {{}, {}, {}, {}};
  int brow[4];
#pragma unroll
  for (int s = 0; s < 4; ++s) {
    int r = tileN + s * 16 + l16;
    brow[s] = (r < N) ? r : (N - 1);   // clamp: OOB subtile acc is never stored
  }

  for (int k0 = 0; k0 < K; k0 += 32) {
    const int kkg = k0 + 8 * g;
    v16bf afrag = load_frag_a(A, K, arow, kkg);
    // near-scope prefetch of the next A chunk (global_prefetch_b8)
    __builtin_prefetch(A + (size_t)arow * K + kkg + 32, 0, 3);
#pragma unroll
    for (int s = 0; s < 4; ++s) {
      v16bf bfrag = load_frag_b(Bt, K, brow[s], k0, g);
      acc[s] = __builtin_amdgcn_wmma_f32_16x16x32_bf16(
          false, afrag, false, bfrag, (short)0, acc[s], false, false);
    }
  }

  const int mbase = tileM + g * 8;   // lanes 0-15: rows 0-7, lanes 16-31: 8-15
#pragma unroll
  for (int s = 0; s < 4; ++s) {
    const int n = tileN + s * 16 + l16;
    if ((tileN + s * 16) >= N) continue;   // wave-uniform (N % 16 == 0)
#pragma unroll
    for (int r = 0; r < 8; ++r) {
      size_t idx = (size_t)(mbase + r) * ldc + n;
      float v = acc[s][r];
      if (Resid) v += Resid[idx];
      Cout[idx] = v;
    }
  }
}

// ---------------------------------------------------------------------------
// fp32 -> bf16 transpose-cast:  Wt[n*K + k] = W[k*N + n]
// ---------------------------------------------------------------------------
__global__ void transpose_cast_bf16(const float* __restrict__ W,
                                    bf16_t* __restrict__ Wt, int K, int N) {
  size_t idx = (size_t)blockIdx.x * blockDim.x + threadIdx.x;
  if (idx >= (size_t)K * N) return;
  int    k = (int)(idx % K);
  size_t n = idx / K;
  Wt[n * K + k] = to_bf16(W[k * (size_t)N + n]);
}

// ---------------------------------------------------------------------------
// RMSNorm -> bf16 rows.  One wave (32 lanes) per token row of D=256.
// ---------------------------------------------------------------------------
__global__ __launch_bounds__(256)
void rmsnorm_bf16(const float* __restrict__ X, const float* __restrict__ w,
                  bf16_t* __restrict__ Out) {
  const int D = DMODEL;
  size_t row = (size_t)blockIdx.x * 8 + (threadIdx.x >> 5);
  int lane = threadIdx.x & 31;
  const float* xr = X + row * D;
  float ss = 0.f;
#pragma unroll
  for (int i = lane; i < D; i += 32) { float v = xr[i]; ss += v * v; }
#pragma unroll
  for (int off = 16; off; off >>= 1) ss += __shfl_xor(ss, off, 32);
  float scale = rsqrtf(ss / (float)D + 1e-5f);
  bf16_t* orow = Out + row * D;
#pragma unroll
  for (int i = lane; i < D; i += 32) orow[i] = to_bf16(xr[i] * scale * w[i]);
}

// ---------------------------------------------------------------------------
// Causal depthwise conv (DCONV=4) + SiLU.  xi lives in first half of xz rows
// (stride 2*ED).  Writes fp32 (for the scan) and bf16 (for x_proj GEMM).
// ---------------------------------------------------------------------------
__global__ void conv_silu(const float* __restrict__ XZ, const float* __restrict__ W,
                          const float* __restrict__ Bias, float* __restrict__ Xc,
                          bf16_t* __restrict__ Xcbf) {
  size_t idx = (size_t)blockIdx.x * blockDim.x + threadIdx.x;
  if (idx >= (size_t)MROWS * ED) return;
  int    e  = (int)(idx % ED);
  size_t bt = idx / ED;
  int    t  = (int)(bt % SEQ);
  size_t b  = bt / SEQ;
  const float* base = XZ + (b * SEQ) * (size_t)(2 * ED) + e;
  float acc = Bias[e];
#pragma unroll
  for (int j = 0; j < DCONV; ++j) {
    int tt = t - (DCONV - 1) + j;
    if (tt >= 0) acc += W[e * DCONV + j] * base[(size_t)tt * (2 * ED)];
  }
  float y = acc / (1.f + __expf(-acc));   // silu
  Xc[idx]   = y;
  Xcbf[idx] = to_bf16(y);
}

// ---------------------------------------------------------------------------
// delta = softplus(dbc[:, :16] @ dt_w + dt_b)   (K=16, plain FMA)
// ---------------------------------------------------------------------------
__global__ void dtproj(const float* __restrict__ DBC, const float* __restrict__ DTW,
                       const float* __restrict__ DTB, float* __restrict__ Delta) {
  size_t idx = (size_t)blockIdx.x * blockDim.x + threadIdx.x;
  if (idx >= (size_t)MROWS * ED) return;
  int    e = (int)(idx % ED);
  size_t m = idx / ED;
  const float* row = DBC + m * (size_t)(DTRANK + 2 * NSTATE);
  float acc = DTB[e];
#pragma unroll
  for (int r = 0; r < DTRANK; ++r) acc += row[r] * DTW[r * ED + e];
  Delta[idx] = (acc > 20.f) ? acc : log1pf(__expf(acc));
}

// ---------------------------------------------------------------------------
// Selective scan.  Thread = (b, e) recurrence; 16 states in registers.
// B/C (shared across all e for a given (b,t)) staged through LDS in chunks
// of TCHUNK timesteps using gfx1250 async global->LDS copies when available
// (ASYNCcnt path), otherwise plain cooperative loads.
// Y may alias Delta (Delta[idx] is read before Y[idx] is written).
// ---------------------------------------------------------------------------
#define TCHUNK 64
__global__ __launch_bounds__(128)
void scan_kernel(const float* __restrict__ Delta, const float* __restrict__ Xc,
                 const float* __restrict__ DBC, const float* __restrict__ Alog,
                 const float* __restrict__ Dp, float* __restrict__ Y) {
  const int b = blockIdx.x;
  const int e = blockIdx.y * 128 + threadIdx.x;
  // per timestep: 16 B values then 16 C values, contiguous (128 bytes)
  __shared__ float sBC[TCHUNK * 32];

  float A[NSTATE], h[NSTATE];
#pragma unroll
  for (int n = 0; n < NSTATE; ++n) {
    A[n] = -__expf(Alog[e * NSTATE + n]);
    h[n] = 0.f;
  }
  const float dpe = Dp[e];
  const size_t baseBT = (size_t)b * SEQ;

  for (int t0 = 0; t0 < SEQ; t0 += TCHUNK) {
    // Stage B/C chunk: TCHUNK rows x 32 floats (cols 16..47 of each DBC row),
    // moved as 16-byte quads.  Row stride 192B keeps every quad 16B-aligned.
#if defined(__gfx1250__) && __has_builtin(__builtin_amdgcn_global_load_async_to_lds_b128)
    for (int i = threadIdx.x; i < TCHUNK * 8; i += 128) {
      int tt = i >> 3, part = i & 7;
      const float* gsrc =
          DBC + (baseBT + t0 + tt) * (size_t)(DTRANK + 2 * NSTATE) + DTRANK + part * 4;
      __builtin_amdgcn_global_load_async_to_lds_b128(
          (__attribute__((address_space(1))) v4i_vs*)gsrc,
          (__attribute__((address_space(3))) v4i_vs*)&sBC[tt * 32 + part * 4],
          0, 0);
    }
#if __has_builtin(__builtin_amdgcn_s_wait_asynccnt)
    __builtin_amdgcn_s_wait_asynccnt(0);
#else
    asm volatile("s_wait_asynccnt 0x0" ::: "memory");
#endif
#else
    for (int i = threadIdx.x; i < TCHUNK * 8; i += 128) {
      int tt = i >> 3, part = i & 7;
      const float4* gsrc = (const float4*)(
          DBC + (baseBT + t0 + tt) * (size_t)(DTRANK + 2 * NSTATE) + DTRANK + part * 4);
      *(float4*)&sBC[tt * 32 + part * 4] = *gsrc;
    }
#endif
    __syncthreads();

#pragma unroll 4
    for (int t = 0; t < TCHUNK; ++t) {
      size_t idx = (baseBT + t0 + t) * (size_t)ED + e;
      float d  = Delta[idx];
      float x  = Xc[idx];
      float dx = d * x;
      float y  = 0.f;
#pragma unroll
      for (int n = 0; n < NSTATE; ++n) {
        float dA = __expf(d * A[n]);
        h[n] = dA * h[n] + dx * sBC[t * 32 + n];
        y   += h[n] * sBC[t * 32 + 16 + n];
      }
      Y[idx] = y + dpe * x;
    }
    __syncthreads();
  }
}

// ---------------------------------------------------------------------------
// y *= silu(z); emit bf16 for out_proj GEMM.  z is the back half of xz rows.
// ---------------------------------------------------------------------------
__global__ void gate_silu(const float* __restrict__ Yin, const float* __restrict__ XZ,
                          bf16_t* __restrict__ Ybf) {
  size_t idx = (size_t)blockIdx.x * blockDim.x + threadIdx.x;
  if (idx >= (size_t)MROWS * ED) return;
  int    e  = (int)(idx % ED);
  size_t bt = idx / ED;
  float z = XZ[bt * (size_t)(2 * ED) + ED + e];
  float gz = z / (1.f + __expf(-z));
  Ybf[idx] = to_bf16(Yin[idx] * gz);
}

// ---------------------------------------------------------------------------
// Mean-pool over SEQ (coalesced: consecutive threads = consecutive d).
// ---------------------------------------------------------------------------
__global__ __launch_bounds__(256)
void pool_kernel(const float* __restrict__ H, float* __restrict__ P) {
  int b = blockIdx.x;
  int d = threadIdx.x;
  float s = 0.f;
  for (int t = 0; t < SEQ; ++t)
    s += H[((size_t)b * SEQ + t) * DMODEL + d];
  P[b * DMODEL + d] = s * (1.f / SEQ);
}

// ---------------------------------------------------------------------------
// head: out[b,o] = pooled[b,:] @ head_w + head_b
// ---------------------------------------------------------------------------
__global__ void head_kernel(const float* __restrict__ P, const float* __restrict__ Hw,
                            const float* __restrict__ Hb, float* __restrict__ Out) {
  int b = blockIdx.x;
  int o = threadIdx.x;   // 64
  float s = Hb[o];
#pragma unroll 8
  for (int d = 0; d < DMODEL; ++d) s += P[b * DMODEL + d] * Hw[d * 64 + o];
  Out[b * 64 + o] = s;
}

// ---------------------------------------------------------------------------
// Host orchestration
// ---------------------------------------------------------------------------
extern "C" void kernel_launch(void* const* d_in, const int* in_sizes, int n_in,
                              void* d_out, int out_size, void* d_ws, size_t ws_size,
                              hipStream_t stream) {
  const float* x      = (const float*)d_in[0];
  const float* rms_w  = (const float*)d_in[1];
  const float* in_w   = (const float*)d_in[2];
  const float* conv_w = (const float*)d_in[3];
  const float* conv_b = (const float*)d_in[4];
  const float* xproj_w= (const float*)d_in[5];
  const float* dt_w   = (const float*)d_in[6];
  const float* dt_b   = (const float*)d_in[7];
  const float* A_log  = (const float*)d_in[8];
  const float* Dp     = (const float*)d_in[9];
  const float* out_w  = (const float*)d_in[10];
  const float* head_w = (const float*)d_in[11];
  const float* head_b = (const float*)d_in[12];
  float* out = (float*)d_out;

  const size_t M = MROWS;
  char* ws = (char*)d_ws;
  size_t o = 0;
  auto alloc = [&](size_t bytes) { size_t r = o; o += (bytes + 255) & ~(size_t)255; return r; };

  size_t off_h     = alloc(M * DMODEL * 4);       // residual stream (fp32)
  size_t off_xnbf  = alloc(M * DMODEL * 2);       // rmsnorm out (bf16)
  size_t off_xz    = alloc(M * 2 * ED * 4);       // in_proj out (fp32)
  size_t off_xc    = alloc(M * ED * 4);           // conv+silu out (fp32)
  size_t off_xcbf  = alloc(M * ED * 2);           // conv out (bf16)
  size_t off_dbc   = alloc(M * (DTRANK + 2 * NSTATE) * 4);
  size_t off_delta = alloc(M * ED * 4);           // delta, then scan y in place
  size_t off_ybf   = alloc(M * ED * 2);           // gated y (bf16)
  size_t off_inwt  = alloc((size_t)(2 * ED) * DMODEL * 2);
  size_t off_xpwt  = alloc((size_t)(DTRANK + 2 * NSTATE) * ED * 2);
  size_t off_outwt = alloc((size_t)DMODEL * ED * 2);
  size_t off_pool  = alloc((size_t)BATCH * DMODEL * 4);

  float*  h     = (float*)(ws + off_h);
  bf16_t* xnbf  = (bf16_t*)(ws + off_xnbf);
  float*  xz    = (float*)(ws + off_xz);
  float*  xc    = (float*)(ws + off_xc);
  bf16_t* xcbf  = (bf16_t*)(ws + off_xcbf);
  float*  dbc   = (float*)(ws + off_dbc);
  float*  delta = (float*)(ws + off_delta);
  bf16_t* ybf   = (bf16_t*)(ws + off_ybf);
  bf16_t* inwt  = (bf16_t*)(ws + off_inwt);
  bf16_t* xpwt  = (bf16_t*)(ws + off_xpwt);
  bf16_t* outwt = (bf16_t*)(ws + off_outwt);
  float*  pooled= (float*)(ws + off_pool);

  // residual stream h = x
  (void)hipMemcpyAsync(h, x, M * DMODEL * sizeof(float), hipMemcpyDeviceToDevice, stream);

  const int EW = 256;                         // elementwise block
  const size_t nME = M * ED;
  const int gME = (int)((nME + EW - 1) / EW);

  for (int l = 0; l < NLAYERS; ++l) {
    const float* rms_l  = rms_w  + (size_t)l * DMODEL;
    const float* inw_l  = in_w   + (size_t)l * DMODEL * (2 * ED);
    const float* cw_l   = conv_w + (size_t)l * ED * DCONV;
    const float* cb_l   = conv_b + (size_t)l * ED;
    const float* xpw_l  = xproj_w+ (size_t)l * ED * (DTRANK + 2 * NSTATE);
    const float* dtw_l  = dt_w   + (size_t)l * DTRANK * ED;
    const float* dtb_l  = dt_b   + (size_t)l * ED;
    const float* Al_l   = A_log  + (size_t)l * ED * NSTATE;
    const float* Dp_l   = Dp     + (size_t)l * ED;
    const float* ow_l   = out_w  + (size_t)l * ED * DMODEL;

    // weight transpose-casts (fp32 -> bf16, [N,K] layout)
    {
      size_t n1 = (size_t)DMODEL * (2 * ED);
      transpose_cast_bf16<<<(int)((n1 + EW - 1) / EW), EW, 0, stream>>>(inw_l, inwt, DMODEL, 2 * ED);
      size_t n2 = (size_t)ED * (DTRANK + 2 * NSTATE);
      transpose_cast_bf16<<<(int)((n2 + EW - 1) / EW), EW, 0, stream>>>(xpw_l, xpwt, ED, DTRANK + 2 * NSTATE);
      size_t n3 = (size_t)ED * DMODEL;
      transpose_cast_bf16<<<(int)((n3 + EW - 1) / EW), EW, 0, stream>>>(ow_l, outwt, ED, DMODEL);
    }

    // rmsnorm -> bf16
    rmsnorm_bf16<<<(int)(M / 8), 256, 0, stream>>>(h, rms_l, xnbf);

    // xz = xn @ in_w           [M,1024] = [M,256]x[256,1024]
    gemm_bf16_wmma<<<dim3((2 * ED + 127) / 128, (int)(M / 64)), 256, 0, stream>>>(
        xnbf, inwt, xz, nullptr, (int)M, 2 * ED, DMODEL, 2 * ED);

    // causal conv + silu
    conv_silu<<<gME, EW, 0, stream>>>(xz, cw_l, cb_l, xc, xcbf);

    // dbc = xc @ xproj_w       [M,48] = [M,512]x[512,48]
    gemm_bf16_wmma<<<dim3(1, (int)(M / 64)), 256, 0, stream>>>(
        xcbf, xpwt, dbc, nullptr, (int)M, DTRANK + 2 * NSTATE, ED, DTRANK + 2 * NSTATE);

    // delta = softplus(dt @ dt_w + dt_b)
    dtproj<<<gME, EW, 0, stream>>>(dbc, dtw_l, dtb_l, delta);

    // selective scan (y written over delta in place)
    scan_kernel<<<dim3(BATCH, ED / 128), 128, 0, stream>>>(delta, xc, dbc, Al_l, Dp_l, delta);

    // y *= silu(z) -> bf16
    gate_silu<<<gME, EW, 0, stream>>>(delta, xz, ybf);

    // h += y @ out_w           [M,256] += [M,512]x[512,256]
    gemm_bf16_wmma<<<dim3((DMODEL + 127) / 128, (int)(M / 64)), 256, 0, stream>>>(
        ybf, outwt, h, h, (int)M, DMODEL, ED, DMODEL);
  }

  // mean pool + head
  pool_kernel<<<BATCH, DMODEL, 0, stream>>>(h, pooled);
  head_kernel<<<BATCH, 64, 0, stream>>>(pooled, head_w, head_b, out);

  (void)in_sizes; (void)n_in; (void)out_size; (void)ws_size;
}
